// _fasterRCNN_10943576670543
// MI455X (gfx1250) — compile-verified, matrix operand order checked by reading, not run on hardware
//
#include <hip/hip_runtime.h>
#include <hip/hip_bf16.h>
#include <math.h>

typedef __attribute__((ext_vector_type(16))) __bf16 v16bf;
typedef __attribute__((ext_vector_type(8)))  __bf16 v8bf;
typedef __attribute__((ext_vector_type(8)))  float  v8f;

#define G 9
#define N 4096
#define D_ORI 2048
#define D_FEAT 512
#define NCLS 21
#define W2COLS 32     // 4 reg + 21 cls + 7 pad

#define KSTEP 64      // staged K per iteration (2 WMMA k-chunks of 32)
#define APAD  8       // LDS row padding (bf16) -> stride 72 = 36 dwords (conflict-free)
#define ASTRIDE (KSTEP + APAD)

// ---------------- workspace layout (bytes) ----------------
#define WS_ATT_T   ((size_t)0)                                   // bf16 [512][2048]
#define WS_ATT_SZ  ((size_t)D_FEAT * D_ORI * 2)
#define WS_W2T     (WS_ATT_T + WS_ATT_SZ)                        // bf16 [G][32][2048]
#define WS_W2_SZ   ((size_t)G * W2COLS * D_ORI * 2)
#define WS_FOFF    (WS_W2T + WS_W2_SZ)                           // bf16 [G][N][512]
#define WS_FOFF_SZ ((size_t)G * N * D_FEAT * 2)
#define WS_CLSREG  (WS_FOFF + WS_FOFF_SZ)                        // f32 [G][N][32]
#define WS_CR_SZ   ((size_t)G * N * W2COLS * 4)
#define WS_ALPHA   (WS_CLSREG + WS_CR_SZ)                        // f32 [G][N][4]
#define WS_AL_SZ   ((size_t)G * N * 4 * 4)
#define WS_ACLS    (WS_ALPHA + WS_AL_SZ)                         // f32 [G][N]
#define WS_AC_SZ   ((size_t)G * N * 4)

static __device__ inline v16bf cat16(v8bf lo, v8bf hi) {
    v16bf r;
#pragma unroll
    for (int i = 0; i < 8; ++i) { r[i] = lo[i]; r[i + 8] = hi[i]; }
    return r;
}

static __device__ inline v8f wmma_bf16(v16bf a, v16bf b, v8f c) {
    // D = A(16x32 bf16) * B(32x16 bf16) + C(16x16 f32)
    return __builtin_amdgcn_wmma_f32_16x16x32_bf16(
        /*neg_a=*/false, a, /*neg_b=*/false, b,
        /*c_mod=*/(short)0, c, /*reuse_a=*/false, /*reuse_b=*/false);
}

// ---- kernel 1a: attention_feat [2048,512] f32 -> attT bf16 [512][2048] ----
__global__ __launch_bounds__(256) void k_att_t(const float* __restrict__ att,
                                               __bf16* __restrict__ attT) {
    int idx = blockIdx.x * 256 + threadIdx.x;     // 2048*512 elems, n fastest
    int k = idx >> 9;
    int n = idx & 511;
    attT[(size_t)n * D_ORI + k] = (__bf16)att[idx];
}

// ---- kernel 1b: pack [bbox_regress | cls_w | 0] -> w2T bf16 [G][32][2048] ----
__global__ __launch_bounds__(256) void k_w2t(const float* __restrict__ bbox,
                                             const float* __restrict__ clsw,
                                             __bf16* __restrict__ w2T) {
    int idx = blockIdx.x * 256 + threadIdx.x;     // G*32*2048
    int g   = idx >> 16;                          // 32*2048 = 65536
    int rem = idx & 65535;
    int j   = rem >> 11;
    int k   = rem & 2047;
    float v = 0.0f;
    if (j < 4)        v = bbox[((size_t)g * D_ORI + k) * 4 + j];
    else if (j < 25)  v = clsw[((size_t)g * D_ORI + k) * NCLS + (j - 4)];
    w2T[idx] = (__bf16)v;
}

// ---- kernel 2: fused main GEMM ----
// grid (64, 9), block 512 (16 wave32s). WG tile: 64 rows x (512 + 32) cols.
// Each wave: 64x32 main cols. W2 (cls+reg, 32 cols) is spread over waves 0..7:
// wave w owns (tile = w&1, row-tile = w>>1); the wave re-reads its own A
// fragment from LDS at a wave-uniform ADDRESS (no register indexing),
// 1 extra WMMA per k-chunk, single v8f accumulator, disjoint epilogue writes.
// Double-buffered LDS A-tile; A, B AND W prefetched one iteration ahead.
__global__ __launch_bounds__(512) void k_main(const float* __restrict__ feat,
                                              const __bf16* __restrict__ attT,
                                              const __bf16* __restrict__ w2T,
                                              __bf16* __restrict__ featOff,
                                              float* __restrict__ clsreg) {
    __shared__ __align__(16) __bf16 As[2][64 * ASTRIDE];  // 2 x 9 KB

    const int g    = blockIdx.y;
    const int n0   = blockIdx.x * 64;
    const int tid  = threadIdx.x;
    const int wave = tid >> 5;
    const int lane = tid & 31;
    const int col0 = wave * 32;
    const int lanelo = lane & 15;
    const int khalf  = lane >> 4;          // A-frag K-half select per ISA layout
    const int wrt    = (wave >> 1) & 3;    // W2 row-tile owned by this wave (waves 0..7)
    // wave-uniform LDS offset of this wave's W2 A-fragment (address select, not reg select)
    const int woff   = (wrt * 16 + lanelo) * ASTRIDE + khalf * 8;

    const v8f zero8 = {0.f, 0.f, 0.f, 0.f, 0.f, 0.f, 0.f, 0.f};
    v8f acc[4][2];
    v8f accw = zero8;
#pragma unroll
    for (int rt = 0; rt < 4; ++rt) { acc[rt][0] = zero8; acc[rt][1] = zero8; }

    const float* Abase = feat + ((size_t)g * N + n0) * D_ORI;
    const __bf16* Bb0 = attT + (size_t)(col0 + lanelo) * D_ORI + khalf * 8;
    const __bf16* Bb1 = attT + (size_t)(col0 + 16 + lanelo) * D_ORI + khalf * 8;
    // (wave & 1) keeps this address valid for ALL waves -> loads stay unconditional
    const __bf16* Wb  = w2T + ((size_t)g * W2COLS + (wave & 1) * 16 + lanelo) * D_ORI + khalf * 8;

    // cooperative A-load: 64 rows x 64 k fp32 per iteration, 8 floats / thread
    const int ldr = tid >> 3;            // 0..63
    const int ldk = (tid & 7) * 8;       // 0,8,...,56
    const float* Aptr = Abase + (size_t)ldr * D_ORI + ldk;

    // ---- prologue: prefetch iteration 0 (A, B, W all to regs) ----
    float4 pf0 = *(const float4*)(Aptr);
    float4 pf1 = *(const float4*)(Aptr + 4);
    v16bf b0c0 = cat16(*(const v8bf*)(Bb0),      *(const v8bf*)(Bb0 + 16));
    v16bf b1c0 = cat16(*(const v8bf*)(Bb1),      *(const v8bf*)(Bb1 + 16));
    v16bf b0c1 = cat16(*(const v8bf*)(Bb0 + 32), *(const v8bf*)(Bb0 + 48));
    v16bf b1c1 = cat16(*(const v8bf*)(Bb1 + 32), *(const v8bf*)(Bb1 + 48));
    v16bf bwc0 = cat16(*(const v8bf*)(Wb),       *(const v8bf*)(Wb + 16));
    v16bf bwc1 = cat16(*(const v8bf*)(Wb + 32),  *(const v8bf*)(Wb + 48));

    constexpr int NIT = D_ORI / KSTEP;   // 32 iterations
#pragma unroll 2
    for (int it = 0; it < NIT; ++it) {
        // convert staged tile -> bf16 LDS (single b128 store)
        {
            v8bf v;
            v[0] = (__bf16)pf0.x; v[1] = (__bf16)pf0.y;
            v[2] = (__bf16)pf0.z; v[3] = (__bf16)pf0.w;
            v[4] = (__bf16)pf1.x; v[5] = (__bf16)pf1.y;
            v[6] = (__bf16)pf1.z; v[7] = (__bf16)pf1.w;
            *(v8bf*)(&As[it & 1][ldr * ASTRIDE + ldk]) = v;
        }
        __syncthreads();

        // clamped next-iteration index: keeps last-iteration prefetch in-bounds
        const int kn = (it + 1 < NIT ? it + 1 : it) * KSTEP;

        // prefetch next iteration's A tile (overlaps all compute below)
        {
            const float* p = Aptr + kn;
            pf0 = *(const float4*)(p);
            pf1 = *(const float4*)(p + 4);
        }
        // prefetch next iteration's B and W fragments (consumed next iteration;
        // unconditional + live across the back-edge, so they stay in the top clause)
        const v16bf nb0c0 = cat16(*(const v8bf*)(Bb0 + kn),      *(const v8bf*)(Bb0 + kn + 16));
        const v16bf nb1c0 = cat16(*(const v8bf*)(Bb1 + kn),      *(const v8bf*)(Bb1 + kn + 16));
        const v16bf nb0c1 = cat16(*(const v8bf*)(Bb0 + kn + 32), *(const v8bf*)(Bb0 + kn + 48));
        const v16bf nb1c1 = cat16(*(const v8bf*)(Bb1 + kn + 32), *(const v8bf*)(Bb1 + kn + 48));
        const v16bf nbwc0 = cat16(*(const v8bf*)(Wb + kn),       *(const v8bf*)(Wb + kn + 16));
        const v16bf nbwc1 = cat16(*(const v8bf*)(Wb + kn + 32),  *(const v8bf*)(Wb + kn + 48));

        const __bf16* lds = &As[it & 1][0];
        // ---- k-chunk 0 ----
        // A frag layout: lane<16 -> K{0-7,16-23}, lane>=16 -> K{8-15,24-31}
        v16bf a[4];
#pragma unroll
        for (int rt = 0; rt < 4; ++rt) {
            const __bf16* p = lds + (rt * 16 + lanelo) * ASTRIDE + khalf * 8;
            a[rt] = cat16(*(const v8bf*)p, *(const v8bf*)(p + 16));
        }
#pragma unroll
        for (int rt = 0; rt < 4; ++rt) {
            acc[rt][0] = wmma_bf16(a[rt], b0c0, acc[rt][0]);
            acc[rt][1] = wmma_bf16(a[rt], b1c0, acc[rt][1]);
        }
        if (wave < 8) {   // wave-uniform: EXEC all-ones inside; address-based frag select
            const __bf16* p = lds + woff;
            const v16bf awf = cat16(*(const v8bf*)p, *(const v8bf*)(p + 16));
            accw = wmma_bf16(awf, bwc0, accw);
        }
        // ---- k-chunk 1 ----
#pragma unroll
        for (int rt = 0; rt < 4; ++rt) {
            const __bf16* p = lds + (rt * 16 + lanelo) * ASTRIDE + 32 + khalf * 8;
            a[rt] = cat16(*(const v8bf*)p, *(const v8bf*)(p + 16));
        }
#pragma unroll
        for (int rt = 0; rt < 4; ++rt) {
            acc[rt][0] = wmma_bf16(a[rt], b0c1, acc[rt][0]);
            acc[rt][1] = wmma_bf16(a[rt], b1c1, acc[rt][1]);
        }
        if (wave < 8) {
            const __bf16* p = lds + woff + 32;
            const v16bf awf = cat16(*(const v8bf*)p, *(const v8bf*)(p + 16));
            accw = wmma_bf16(awf, bwc1, accw);
        }
        // rotate double-buffers (renamed away by the 2x unroll)
        b0c0 = nb0c0; b1c0 = nb1c0; b0c1 = nb0c1; b1c1 = nb1c1;
        bwc0 = nbwc0; bwc1 = nbwc1;
        // no trailing barrier: next iteration writes the OTHER LDS buffer,
        // and its leading barrier orders the store against this compute.
    }

    // epilogue: C layout — VGPR v, lane<16: M=v, lane>=16: M=v+8; N=lane&15
#pragma unroll
    for (int rt = 0; rt < 4; ++rt) {
#pragma unroll
        for (int ct = 0; ct < 2; ++ct) {
#pragma unroll
            for (int v = 0; v < 8; ++v) {
                float x = acc[rt][ct][v];
                x = x > 0.f ? x : 0.f;                       // relu
                const int row = n0 + rt * 16 + v + 8 * khalf;
                const int col = col0 + ct * 16 + lanelo;
                featOff[((size_t)g * N + row) * D_FEAT + col] = (__bf16)x;
            }
        }
    }
    if (wave < 8) {   // disjoint (row-tile, col-tile) per wave: no reduction needed
#pragma unroll
        for (int v = 0; v < 8; ++v) {
            const int row = n0 + wrt * 16 + v + 8 * khalf;
            const int col = (wave & 1) * 16 + lanelo;
            clsreg[((size_t)g * N + row) * W2COLS + col] = accw[v];
        }
    }
}

// ---- kernel 3a: alpha/alpha_cls GEMV: wave per (g,n) ----
__global__ __launch_bounds__(256) void k_alpha(const __bf16* __restrict__ featOff,
                                               const float* __restrict__ aw,
                                               const float* __restrict__ awc,
                                               float* __restrict__ alpha,
                                               float* __restrict__ acls) {
    const int wave = threadIdx.x >> 5;
    const int lane = threadIdx.x & 31;
    const int idx  = blockIdx.x * 8 + wave;     // 0 .. G*N-1
    const int g    = idx >> 12;                 // N = 4096
    const __bf16* f = featOff + (size_t)idx * D_FEAT + lane * 16;
    const v8bf f0 = *(const v8bf*)f;
    const v8bf f1 = *(const v8bf*)(f + 8);
    const float* w  = aw  + ((size_t)g * D_FEAT + lane * 16) * 4;
    const float* wc = awc + (size_t)g * D_FEAT + lane * 16;
    float s0 = 0.f, s1 = 0.f, s2 = 0.f, s3 = 0.f, sc = 0.f;
#pragma unroll
    for (int e = 0; e < 8; ++e) {
        const float fv = (float)f0[e];
        const float* w4 = w + e * 4;
        s0 += fv * w4[0]; s1 += fv * w4[1]; s2 += fv * w4[2]; s3 += fv * w4[3];
        sc += fv * wc[e];
    }
#pragma unroll
    for (int e = 0; e < 8; ++e) {
        const float fv = (float)f1[e];
        const float* w4 = w + (8 + e) * 4;
        s0 += fv * w4[0]; s1 += fv * w4[1]; s2 += fv * w4[2]; s3 += fv * w4[3];
        sc += fv * wc[8 + e];
    }
#pragma unroll
    for (int m = 16; m >= 1; m >>= 1) {
        s0 += __shfl_xor(s0, m, 32);
        s1 += __shfl_xor(s1, m, 32);
        s2 += __shfl_xor(s2, m, 32);
        s3 += __shfl_xor(s3, m, 32);
        sc += __shfl_xor(sc, m, 32);
    }
    if (lane == 0) {
        float* ap = alpha + (size_t)idx * 4;
        ap[0] = s0; ap[1] = s1; ap[2] = s2; ap[3] = s3;
        acls[idx] = sc;
    }
}

// ---- kernel 3b: softmaxes + combine; thread per n ----
__global__ __launch_bounds__(256) void k_combine(const float* __restrict__ alpha,
                                                 const float* __restrict__ acls,
                                                 const float* __restrict__ clsreg,
                                                 const float* __restrict__ drois,
                                                 float* __restrict__ out_output,
                                                 float* __restrict__ out_dpo,
                                                 float* __restrict__ out_cls,
                                                 float* __restrict__ out_acls) {
    const int n = blockIdx.x * 256 + threadIdx.x;     // 0..N-1
    float a[G][4], ac[G];
#pragma unroll
    for (int g = 0; g < G; ++g) {
        const float* p = alpha + ((size_t)g * N + n) * 4;
        a[g][0] = p[0]; a[g][1] = p[1]; a[g][2] = p[2]; a[g][3] = p[3];
        ac[g] = acls[(size_t)g * N + n];
    }
    // softmax over g per j; combine with delta_rois + offset
    float outv[4];
#pragma unroll
    for (int j = 0; j < 4; ++j) {
        float mx = a[0][j];
#pragma unroll
        for (int g = 1; g < G; ++g) mx = fmaxf(mx, a[g][j]);
        float e[G], sum = 0.f;
#pragma unroll
        for (int g = 0; g < G; ++g) { e[g] = expf(a[g][j] - mx); sum += e[g]; }
        const float inv = 1.f / sum;
        float o = 0.f;
#pragma unroll
        for (int g = 0; g < G; ++g) {
            const float off = clsreg[((size_t)g * N + n) * W2COLS + j];
            const float dpo = drois[((size_t)g * N + n) * 5 + 1 + j] + off;
            out_dpo[((size_t)g * N + n) * 4 + j] = dpo;
            o += dpo * e[g] * inv;
        }
        outv[j] = o;
    }
#pragma unroll
    for (int j = 0; j < 4; ++j) out_output[(size_t)n * 4 + j] = outv[j];
    // alpha_cls softmax over g
    {
        float mx = ac[0];
#pragma unroll
        for (int g = 1; g < G; ++g) mx = fmaxf(mx, ac[g]);
        float sum = 0.f;
#pragma unroll
        for (int g = 0; g < G; ++g) { ac[g] = expf(ac[g] - mx); sum += ac[g]; }
        const float inv = 1.f / sum;
#pragma unroll
        for (int g = 0; g < G; ++g) out_acls[(size_t)g * N + n] = ac[g] * inv;
    }
    // cls softmax over 21 classes per g
    for (int g = 0; g < G; ++g) {
        const float* c = clsreg + ((size_t)g * N + n) * W2COLS + 4;
        float cv[NCLS], mx = c[0];
#pragma unroll
        for (int i = 0; i < NCLS; ++i) { cv[i] = c[i]; mx = fmaxf(mx, cv[i]); }
        float sum = 0.f;
#pragma unroll
        for (int i = 0; i < NCLS; ++i) { cv[i] = expf(cv[i] - mx); sum += cv[i]; }
        const float inv = 1.f / sum;
        float* dst = out_cls + ((size_t)g * N + n) * NCLS;
#pragma unroll
        for (int i = 0; i < NCLS; ++i) dst[i] = cv[i] * inv;
    }
}

extern "C" void kernel_launch(void* const* d_in, const int* in_sizes, int n_in,
                              void* d_out, int out_size, void* d_ws, size_t ws_size,
                              hipStream_t stream) {
    const float* features   = (const float*)d_in[0];
    // d_in[1] gt_features, d_in[2] rois, d_in[4] ious: unused by reference math
    const float* delta_rois = (const float*)d_in[3];
    const float* att        = (const float*)d_in[5];
    const float* alpha_w    = (const float*)d_in[6];
    const float* bbox       = (const float*)d_in[7];
    const float* awc        = (const float*)d_in[8];
    const float* clsw       = (const float*)d_in[9];

    char* ws = (char*)d_ws;
    __bf16* attT    = (__bf16*)(ws + WS_ATT_T);
    __bf16* w2T     = (__bf16*)(ws + WS_W2T);
    __bf16* featOff = (__bf16*)(ws + WS_FOFF);
    float*  clsreg  = (float*)(ws + WS_CLSREG);
    float*  alphaB  = (float*)(ws + WS_ALPHA);
    float*  aclsB   = (float*)(ws + WS_ACLS);

    float* out_output = (float*)d_out;                  // [N,4]
    float* out_dpo    = out_output + (size_t)N * 4;     // [G,N,4]
    float* out_cls    = out_dpo + (size_t)G * N * 4;    // [G,N,21]
    float* out_acls   = out_cls + (size_t)G * N * NCLS; // [G,N,1]

    k_att_t<<<(D_ORI * D_FEAT) / 256, 256, 0, stream>>>(att, attT);
    k_w2t<<<(G * W2COLS * D_ORI) / 256, 256, 0, stream>>>(bbox, clsw, w2T);
    k_main<<<dim3(N / 64, G), 512, 0, stream>>>(features, attT, w2T, featOff, clsreg);
    k_alpha<<<(G * N) / 8, 256, 0, stream>>>(featOff, alpha_w, awc, alphaB, aclsB);
    k_combine<<<N / 256, 256, 0, stream>>>(alphaB, aclsB, clsreg, delta_rois,
                                           out_output, out_dpo, out_cls, out_acls);
}
